// DGCNN_Transformer_3951369912526
// MI455X (gfx1250) — matrix operand order repaired and checked by reading, not compiled, not run on hardware
//
#include <hip/hip_runtime.h>
#include <hip/hip_bf16.h>

typedef _Float16 v16h __attribute__((ext_vector_type(16)));
typedef _Float16 vh8  __attribute__((ext_vector_type(8)));
typedef _Float16 vh4  __attribute__((ext_vector_type(4)));
typedef _Float16 vh2  __attribute__((ext_vector_type(2)));
typedef float    v8f  __attribute__((ext_vector_type(8)));
typedef float    vf4  __attribute__((ext_vector_type(4)));

static constexpr int B_ = 16, N_ = 2048, KNN = 20, HID_ = 1024, NCLS_ = 16;

// ---------------------------------------------------------------------------
// WMMA GEMM (per batch in blockIdx.z), f16 activations in HBM:
//   Y[o][m] = act( (sum_c W[o][c] * X[b][c][m]) * scale[o] + shift[o] )
// ACT: 1 = relu, 2 = leaky(0.2). scale==null -> 1, shift==null -> 0.
// Requirements (guaranteed by caller): O % 64 == 0, M % 64 == 0.
// Block = 256 threads = 8 waves; block tile 64(O) x 64(M); wave (wo,wm)
// computes two 16x16 sub-tiles per K-step, sharing one A fragment.
// LDS: Wt row-major [64o][32k]; Xt K-transposed [64m][40(k pad)] so every
// lane's A/B fragment is two aligned 16B chunks -> ds_load_b128.
// ---------------------------------------------------------------------------
template<int ACT>
__global__ __launch_bounds__(256)
void gemm_cbr_kernel(const float* __restrict__ W, const _Float16* __restrict__ X,
                     _Float16* __restrict__ Y,
                     const float* __restrict__ scale, const float* __restrict__ shift,
                     int O, int C, int M)
{
    __shared__ alignas(16) _Float16 Wt[64 * 32];     // [o][k]
    __shared__ alignas(16) _Float16 Xt[64 * 40];     // [m][k], stride 40 halves

    const int b      = blockIdx.z;
    const long Mb    = (long)M;
    const _Float16* Xb = X + (long)b * C * Mb;
    _Float16*       Yb = Y + (long)b * O * Mb;
    const int m_base = blockIdx.x * 64;
    const int o_base = blockIdx.y * 64;
    const int tid    = threadIdx.x;
    const int lane   = tid & 31;
    const int wid    = tid >> 5;
    const int wo     = wid >> 1;          // 0..3  (O sub-tile)
    const int wm     = wid & 1;           // 0..1  (M half)
    const int l      = lane & 15;
    const int hi     = lane >> 4;

    // fast-path load indices
    const int wr = tid >> 2, wj = (tid & 3) * 8;     // W: row 0..63, k-offset 0/8/16/24
    const int c2 = tid >> 4, mj = (tid & 15) * 4;    // X: k-pair 0..15, m-offset 0..60

    v8f acc0 = {0.f,0.f,0.f,0.f,0.f,0.f,0.f,0.f};
    v8f acc1 = {0.f,0.f,0.f,0.f,0.f,0.f,0.f,0.f};

    for (int c0 = 0; c0 < C; c0 += 32) {
        if (c0 + 32 <= C) {
            // ---- fast path: unconditional vectorized loads
            {   // W tile (f32): 8 contiguous k per thread -> one ds_store_b128
                const float* wp = W + (long)(o_base + wr) * C + c0 + wj;
                vf4 wa = *(const vf4*)wp;
                vf4 wb = *(const vf4*)(wp + 4);
                vh8 hv;
                #pragma unroll
                for (int i = 0; i < 4; ++i) { hv[i] = (_Float16)wa[i]; hv[4 + i] = (_Float16)wb[i]; }
                *(vh8*)&Wt[wr * 32 + wj] = hv;
            }
            {   // X tile (f16): two k-rows (2*c2, 2*c2+1) x 4 m -> 4 packed b32 LDS stores
                const _Float16* xp = Xb + (long)(c0 + 2 * c2) * Mb + m_base + mj;
                vh4 xa  = *(const vh4*)xp;
                vh4 xb2 = *(const vh4*)(xp + Mb);
                #pragma unroll
                for (int i = 0; i < 4; ++i) {
                    vh2 p; p[0] = xa[i]; p[1] = xb2[i];
                    *(vh2*)&Xt[(mj + i) * 40 + 2 * c2] = p;
                }
            }
            if (c0 + 32 < C)   // prefetch next X tile (global_prefetch_b8)
                __builtin_prefetch(Xb + (long)(c0 + 32 + (tid >> 5)) * Mb + m_base + (tid & 31) * 2, 0, 1);
        } else {
            // ---- K tail (C=3 / C=6 layers): clamped unconditional loads + mask
            #pragma unroll
            for (int i = 0; i < 8; ++i) {
                int idx = tid + i * 256;
                int r = idx >> 5, j = idx & 31;
                int c = c0 + j, cc = c < C ? c : C - 1;
                float v = W[(long)(o_base + r) * C + cc];
                Wt[r * 32 + j] = (_Float16)(c < C ? v : 0.f);
            }
            #pragma unroll
            for (int i = 0; i < 8; ++i) {
                int idx = tid + i * 256;
                int r = idx >> 6, m = idx & 63;          // r = k row 0..31
                int c = c0 + r, cc = c < C ? c : C - 1;
                _Float16 v = Xb[(long)cc * Mb + m_base + m];
                Xt[m * 40 + r] = (c < C) ? v : (_Float16)0.f;
            }
        }
        __syncthreads();

        // ---- fragments: two ds_load_b128 each (contiguous k order, same in A and B)
        // A row = o sub-row; lanes 0-15 hold K 0..15, lanes 16-31 hold K 16..31
        const _Float16* wrow = &Wt[(wo * 16 + l) * 32];
        vh8 a0 = *(const vh8*)(wrow + 16 * hi);
        vh8 a1 = *(const vh8*)(wrow + 16 * hi + 8);
        v16h afrag = __builtin_shufflevector(a0, a1, 0,1,2,3,4,5,6,7,8,9,10,11,12,13,14,15);

        // B col n: lanes 0-15 hold K 0..15, lanes 16-31 hold K 16..31 (contig in Xt row)
        const _Float16* xr0 = &Xt[(wm * 32 + l) * 40 + 16 * hi];
        vh8 b0 = *(const vh8*)xr0;
        vh8 b1 = *(const vh8*)(xr0 + 8);
        v16h bfrag0 = __builtin_shufflevector(b0, b1, 0,1,2,3,4,5,6,7,8,9,10,11,12,13,14,15);

        const _Float16* xr1 = &Xt[(wm * 32 + 16 + l) * 40 + 16 * hi];
        vh8 b2 = *(const vh8*)xr1;
        vh8 b3 = *(const vh8*)(xr1 + 8);
        v16h bfrag1 = __builtin_shufflevector(b2, b3, 0,1,2,3,4,5,6,7,8,9,10,11,12,13,14,15);

        acc0 = __builtin_amdgcn_wmma_f32_16x16x32_f16(false, afrag, false, bfrag0,
                                                      (short)0, acc0, false, false);
        acc1 = __builtin_amdgcn_wmma_f32_16x16x32_f16(false, afrag, false, bfrag1,
                                                      (short)0, acc1, false, false);
        __syncthreads();
    }

    // ---- epilogue. C/D layout: VGPR r: lanes 0-15 -> (M=r, N=lane);
    //                                    lanes 16-31 -> (M=8+r, N=lane-16)
    const int m0 = m_base + wm * 32 + l;
    #pragma unroll
    for (int r = 0; r < 8; ++r) {
        const int o = o_base + wo * 16 + r + 8 * hi;
        const float s = scale ? scale[o] : 1.f;
        const float t = shift ? shift[o] : 0.f;
        float y0 = acc0[r] * s + t;
        float y1 = acc1[r] * s + t;
        if (ACT == 1) { y0 = fmaxf(y0, 0.f);            y1 = fmaxf(y1, 0.f); }
        if (ACT == 2) { y0 = (y0 > 0.f) ? y0 : 0.2f*y0; y1 = (y1 > 0.f) ? y1 : 0.2f*y1; }
        Yb[(long)o * Mb + m0]      = (_Float16)y0;
        Yb[(long)o * Mb + m0 + 16] = (_Float16)y1;
    }
}

// ---------------------------------------------------------------------------
// kNN top-20 of d(n,m) = 2*x_n.x_m - |x_n|^2 - |x_m|^2 (largest = nearest).
// f32 inputs for selection robustness.
// ---------------------------------------------------------------------------
template<int C>
__global__ __launch_bounds__(128)
void knn_kernel(const float* __restrict__ xc, int* __restrict__ idx)
{
    __shared__ float Fown[C * 128];
    __shared__ float Fm[C * 128];
    __shared__ float Sqm[128];

    const int b   = blockIdx.y;
    const int n0  = blockIdx.x * 128;
    const int tid = threadIdx.x;
    const float* xb = xc + (long)b * C * N_;

    for (int i = tid; i < C * 128; i += 128) {
        int c = i >> 7, nl = i & 127;
        Fown[c * 128 + nl] = xb[(long)c * N_ + n0 + nl];
    }
    __syncthreads();

    float sq_own = 0.f;
    #pragma unroll
    for (int c = 0; c < C; ++c) { float v = Fown[c * 128 + tid]; sq_own += v * v; }

    float best[KNN]; int bidx[KNN];
    #pragma unroll
    for (int k = 0; k < KNN; ++k) { best[k] = -1e30f; bidx[k] = 0; }

    for (int m0 = 0; m0 < N_; m0 += 128) {
        __syncthreads();
        for (int i = tid; i < C * 128; i += 128) {
            int c = i >> 7, ml = i & 127;
            Fm[c * 128 + ml] = xb[(long)c * N_ + m0 + ml];
        }
        __syncthreads();
        {
            float s = 0.f;
            #pragma unroll
            for (int c = 0; c < C; ++c) { float v = Fm[c * 128 + tid]; s += v * v; }
            Sqm[tid] = s;
        }
        __syncthreads();

        for (int ml = 0; ml < 128; ++ml) {
            float dot = 0.f;
            #pragma unroll 8
            for (int c = 0; c < C; ++c) dot += Fown[c * 128 + tid] * Fm[c * 128 + ml];
            float d = 2.f * dot - sq_own - Sqm[ml];
            if (d > best[KNN - 1]) {                 // early reject
                float v = d; int vi = m0 + ml;       // branchless sorted insert
                #pragma unroll
                for (int k = 0; k < KNN; ++k) {
                    bool ins = v > best[k];
                    float ov = best[k]; int oi = bidx[k];
                    best[k] = ins ? v : best[k];
                    bidx[k] = ins ? vi : bidx[k];
                    v  = ins ? ov : v;
                    vi = ins ? oi : vi;
                }
            }
        }
    }
    const int n = n0 + tid;
    #pragma unroll
    for (int k = 0; k < KNN; ++k)
        idx[((long)b * N_ + n) * KNN + k] = bidx[k];
}

// ---------------------------------------------------------------------------
// edge (f16 out, GEMM operand): [b, c, n, k] = x[c,idx]-x[c,n] | x[c-C,n]
// ---------------------------------------------------------------------------
__global__ void edge_feature_kernel(const float* __restrict__ xc,
                                    const int* __restrict__ idx,
                                    _Float16* __restrict__ e, int C)
{
    long t = (long)blockIdx.x * blockDim.x + threadIdx.x;
    const long total = (long)B_ * N_ * KNN;
    if (t >= total) return;
    int k = (int)(t % KNN); long bn = t / KNN;
    int n = (int)(bn % N_); int b = (int)(bn / N_);
    int j = idx[t];
    const float* xb = xc + (long)b * C * N_;
    _Float16* eb = e + (long)b * (2 * C) * N_ * KNN;
    for (int c = 0; c < C; ++c) {
        float ctr = xb[(long)c * N_ + n];
        float nbv = xb[(long)c * N_ + j];
        eb[((long)c * N_ + n) * KNN + k]       = (_Float16)(nbv - ctr);
        eb[((long)(c + C) * N_ + n) * KNN + k] = (_Float16)ctr;
    }
}

__global__ void max_k_kernel(const _Float16* __restrict__ e, float* __restrict__ out, int C)
{
    long t = (long)blockIdx.x * blockDim.x + threadIdx.x;
    const long total = (long)B_ * C * N_;
    if (t >= total) return;
    const _Float16* p = e + t * KNN;     // [b][c][n][k] contiguous in k
    float m = (float)p[0];
    #pragma unroll
    for (int k = 1; k < KNN; ++k) m = fmaxf(m, (float)p[k]);
    out[t] = m;
}

__global__ void max_n_kernel(const _Float16* __restrict__ h, float* __restrict__ g, int C)
{
    int t = blockIdx.x * blockDim.x + threadIdx.x;
    if (t >= B_ * C) return;
    const _Float16* p = h + (long)t * N_;
    float m = (float)p[0];
    for (int n = 1; n < N_; ++n) m = fmaxf(m, (float)p[n]);
    g[t] = m;
}

__global__ void xpose_kernel(const float* __restrict__ x, _Float16* __restrict__ xc)
{
    int t = blockIdx.x * blockDim.x + threadIdx.x;
    if (t >= B_ * N_) return;
    int n = t % N_, b = t / N_;
    #pragma unroll
    for (int c = 0; c < 3; ++c)
        xc[((long)b * 3 + c) * N_ + n] = (_Float16)x[((long)b * N_ + n) * 3 + c];
}

__global__ void apply_stn_kernel(const float* __restrict__ x,
                                 const float* __restrict__ trans,
                                 float* __restrict__ xc)
{
    int t = blockIdx.x * blockDim.x + threadIdx.x;
    if (t >= B_ * N_) return;
    int n = t % N_, b = t / N_;
    const float* T = trans + b * 9;
    float x0 = x[((long)b * N_ + n) * 3 + 0];
    float x1 = x[((long)b * N_ + n) * 3 + 1];
    float x2 = x[((long)b * N_ + n) * 3 + 2];
    #pragma unroll
    for (int d = 0; d < 3; ++d)
        xc[((long)b * 3 + d) * N_ + n] = x0 * T[0 * 3 + d] + x1 * T[1 * 3 + d] + x2 * T[2 * 3 + d];
}

__global__ void fc_kernel(const float* __restrict__ W, const float* __restrict__ x,
                          const float* __restrict__ bias, float* __restrict__ y,
                          int O, int C, int act, int add_eye)
{
    int t = blockIdx.x * blockDim.x + threadIdx.x;
    if (t >= B_ * O) return;
    int o = t % O, b = t / O;
    float acc = bias ? bias[o] : 0.f;
    for (int c = 0; c < C; ++c) acc += W[(long)o * C + c] * x[(long)b * C + c];
    if (add_eye && (o == 0 || o == 4 || o == 8)) acc += 1.f;
    if (act) acc = fmaxf(acc, 0.f);
    y[(long)b * O + o] = acc;
}

__global__ void lv_kernel(const float* __restrict__ w7, const float* __restrict__ s,
                          const float* __restrict__ sh, const int* __restrict__ lbl,
                          float* __restrict__ lv)
{
    int t = blockIdx.x * blockDim.x + threadIdx.x;
    if (t >= B_ * 64) return;
    int o = t % 64, b = t / 64;
    float y = w7[o * NCLS_ + lbl[b]] * s[o] + sh[o];
    lv[b * 64 + o] = (y > 0.f) ? y : 0.2f * y;
}

__global__ void cat192_kernel(const float* __restrict__ x1, const float* __restrict__ x2,
                              const float* __restrict__ x3, _Float16* __restrict__ out)
{
    long t = (long)blockIdx.x * blockDim.x + threadIdx.x;
    const long total = (long)B_ * 192 * N_;
    if (t >= total) return;
    int n = (int)(t % N_); long bc = t / N_;
    int c = (int)(bc % 192); int b = (int)(bc / 192);
    float v;
    if (c < 64)       v = x1[((long)b * 64 + c) * N_ + n];
    else if (c < 128) v = x2[((long)b * 64 + (c - 64)) * N_ + n];
    else              v = x3[((long)b * 64 + (c - 128)) * N_ + n];
    out[t] = (_Float16)v;
}

__global__ void cat1280_kernel(const float* __restrict__ g, const float* __restrict__ lv,
                               const float* __restrict__ x1, const float* __restrict__ x2,
                               const float* __restrict__ x3, _Float16* __restrict__ out)
{
    long t = (long)blockIdx.x * blockDim.x + threadIdx.x;
    const long total = (long)B_ * 1280 * N_;
    if (t >= total) return;
    int n = (int)(t % N_); long bc = t / N_;
    int c = (int)(bc % 1280); int b = (int)(bc / 1280);
    float v;
    if (c < 1024)      v = g[(long)b * 1024 + c];
    else if (c < 1088) v = lv[(long)b * 64 + (c - 1024)];
    else if (c < 1152) v = x1[((long)b * 64 + (c - 1088)) * N_ + n];
    else if (c < 1216) v = x2[((long)b * 64 + (c - 1152)) * N_ + n];
    else               v = x3[((long)b * 64 + (c - 1216)) * N_ + n];
    out[t] = (_Float16)v;
}

__global__ void final_kernel(const float* __restrict__ w11, const _Float16* __restrict__ h,
                             float* __restrict__ out)
{
    int t = blockIdx.x * blockDim.x + threadIdx.x;
    if (t >= B_ * N_) return;
    int n = t % N_, b = t / N_;
    #pragma unroll
    for (int o = 0; o < 3; ++o) {
        float acc = 0.f;
        for (int c = 0; c < 128; ++c)
            acc += w11[o * 128 + c] * (float)h[((long)b * 128 + c) * N_ + n];
        out[((long)b * N_ + n) * 3 + o] = tanhf(acc);
    }
}

// ---------------------------------------------------------------------------
static void gemm_cbr(const float* W, const _Float16* X, _Float16* Y,
                     const float* s, const float* sh,
                     int O, int C, int M, int act, hipStream_t stream)
{
    dim3 grid(M / 64, O / 64, B_);
    if (act == 1)
        gemm_cbr_kernel<1><<<grid, 256, 0, stream>>>(W, X, Y, s, sh, O, C, M);
    else
        gemm_cbr_kernel<2><<<grid, 256, 0, stream>>>(W, X, Y, s, sh, O, C, M);
}

extern "C" void kernel_launch(void* const* d_in, const int* in_sizes, int n_in,
                              void* d_out, int out_size, void* d_ws, size_t ws_size,
                              hipStream_t stream)
{
    (void)in_sizes; (void)n_in; (void)out_size; (void)ws_size;
    const float* x        = (const float*)d_in[0];
    const int*   lbl      = (const int*)d_in[1];
    const float* stn_c1_w = (const float*)d_in[2];  const float* stn_c1_b = (const float*)d_in[3];
    const float* stn_c2_w = (const float*)d_in[4];  const float* stn_c2_b = (const float*)d_in[5];
    const float* stn_c3_w = (const float*)d_in[6];  const float* stn_c3_b = (const float*)d_in[7];
    const float* fc1_w    = (const float*)d_in[8];  const float* fc1_b    = (const float*)d_in[9];
    const float* fc2_w    = (const float*)d_in[10]; const float* fc2_b    = (const float*)d_in[11];
    const float* fc3_w    = (const float*)d_in[12]; const float* fc3_b    = (const float*)d_in[13];
    const float* cw[11]; const float* cs[11]; const float* csh[11];
    for (int i = 0; i < 10; ++i) {
        cw[i]  = (const float*)d_in[14 + 3 * i];
        cs[i]  = (const float*)d_in[15 + 3 * i];
        csh[i] = (const float*)d_in[16 + 3 * i];
    }
    const float* conv11_w = (const float*)d_in[44];
    float* out = (float*)d_out;

    // ---- workspace bump allocator
    char* wsb = (char*)d_ws;
    size_t off = 0;
    auto alloc = [&](size_t nbytes) -> void* {
        off = (off + 255) & ~(size_t)255;
        void* p = wsb + off; off += nbytes; return p;
    };
    // f16 GEMM activations
    _Float16* xc0   = (_Float16*)alloc((size_t)B_ * 3 * N_ * 2);
    _Float16* stn1  = (_Float16*)alloc((size_t)B_ * 64 * N_ * 2);
    _Float16* stn2  = (_Float16*)alloc((size_t)B_ * 128 * N_ * 2);
    _Float16* big   = (_Float16*)alloc((size_t)B_ * 1280 * N_ * 2);  // stn_h3 / h6 / cat1280
    _Float16* edge  = (_Float16*)alloc((size_t)B_ * 128 * N_ * KNN * 2);
    _Float16* ecA   = (_Float16*)alloc((size_t)B_ * 64 * N_ * KNN * 2);
    _Float16* ecB   = (_Float16*)alloc((size_t)B_ * 64 * N_ * KNN * 2);
    _Float16* cat19 = (_Float16*)alloc((size_t)B_ * 192 * N_ * 2);
    _Float16* h8    = (_Float16*)alloc((size_t)B_ * 256 * N_ * 2);
    _Float16* h9    = (_Float16*)alloc((size_t)B_ * 256 * N_ * 2);
    _Float16* h10   = (_Float16*)alloc((size_t)B_ * 128 * N_ * 2);
    // f32 buffers (selection / FC / transforms)
    float* g1024 = (float*)alloc((size_t)B_ * 1024 * 4);
    float* fc1o  = (float*)alloc((size_t)B_ * 512 * 4);
    float* fc2o  = (float*)alloc((size_t)B_ * 256 * 4);
    float* trans = (float*)alloc((size_t)B_ * 9 * 4);
    float* xc    = (float*)alloc((size_t)B_ * 3 * N_ * 4);
    float* x1    = (float*)alloc((size_t)B_ * 64 * N_ * 4);
    float* x2    = (float*)alloc((size_t)B_ * 64 * N_ * 4);
    float* x3    = (float*)alloc((size_t)B_ * 64 * N_ * 4);
    float* lv    = (float*)alloc((size_t)B_ * 64 * 4);
    int*   idx   = (int*)  alloc((size_t)B_ * N_ * KNN * 4);

    const int BN = B_ * N_;
    const int NK = N_ * KNN;

    // ---- STN
    xpose_kernel<<<(BN + 255) / 256, 256, 0, stream>>>(x, xc0);
    gemm_cbr(stn_c1_w, xc0,  stn1, nullptr, stn_c1_b,   64,    3, N_, 1, stream);
    gemm_cbr(stn_c2_w, stn1, stn2, nullptr, stn_c2_b,  128,   64, N_, 1, stream);
    gemm_cbr(stn_c3_w, stn2, big,  nullptr, stn_c3_b, 1024,  128, N_, 1, stream);
    max_n_kernel<<<(B_ * 1024 + 255) / 256, 256, 0, stream>>>(big, g1024, 1024);
    fc_kernel<<<(B_ * 512 + 255) / 256, 256, 0, stream>>>(fc1_w, g1024, fc1_b, fc1o, 512, 1024, 1, 0);
    fc_kernel<<<(B_ * 256 + 255) / 256, 256, 0, stream>>>(fc2_w, fc1o,  fc2_b, fc2o, 256,  512, 1, 0);
    fc_kernel<<<(B_ *   9 + 255) / 256, 256, 0, stream>>>(fc3_w, fc2o,  fc3_b, trans,  9,  256, 0, 1);
    apply_stn_kernel<<<(BN + 255) / 256, 256, 0, stream>>>(x, trans, xc);

    // ---- EdgeConv stage 1 (C=3 -> edge 6 -> 64 -> 64 -> x1)
    knn_kernel<3><<<dim3(N_ / 128, B_), 128, 0, stream>>>(xc, idx);
    edge_feature_kernel<<<((long)B_ * NK + 255) / 256, 256, 0, stream>>>(xc, idx, edge, 3);
    gemm_cbr(cw[0], edge, ecA, cs[0], csh[0], 64,  6, NK, 2, stream);
    gemm_cbr(cw[1], ecA,  ecB, cs[1], csh[1], 64, 64, NK, 2, stream);
    max_k_kernel<<<((long)B_ * 64 * N_ + 255) / 256, 256, 0, stream>>>(ecB, x1, 64);

    // ---- EdgeConv stage 2 (x1 -> edge 128 -> 64 -> 64 -> x2)
    knn_kernel<64><<<dim3(N_ / 128, B_), 128, 0, stream>>>(x1, idx);
    edge_feature_kernel<<<((long)B_ * NK + 255) / 256, 256, 0, stream>>>(x1, idx, edge, 64);
    gemm_cbr(cw[2], edge, ecA, cs[2], csh[2], 64, 128, NK, 2, stream);
    gemm_cbr(cw[3], ecA,  ecB, cs[3], csh[3], 64,  64, NK, 2, stream);
    max_k_kernel<<<((long)B_ * 64 * N_ + 255) / 256, 256, 0, stream>>>(ecB, x2, 64);

    // ---- EdgeConv stage 3 (x2 -> edge 128 -> 64 -> x3)
    knn_kernel<64><<<dim3(N_ / 128, B_), 128, 0, stream>>>(x2, idx);
    edge_feature_kernel<<<((long)B_ * NK + 255) / 256, 256, 0, stream>>>(x2, idx, edge, 64);
    gemm_cbr(cw[4], edge, ecA, cs[4], csh[4], 64, 128, NK, 2, stream);
    max_k_kernel<<<((long)B_ * 64 * N_ + 255) / 256, 256, 0, stream>>>(ecA, x3, 64);

    // ---- Global feature
    cat192_kernel<<<((long)B_ * 192 * N_ + 255) / 256, 256, 0, stream>>>(x1, x2, x3, cat19);
    gemm_cbr(cw[5], cat19, big, cs[5], csh[5], 1024, 192, N_, 2, stream);   // conv6
    max_n_kernel<<<(B_ * 1024 + 255) / 256, 256, 0, stream>>>(big, g1024, 1024);
    lv_kernel<<<(B_ * 64 + 255) / 256, 256, 0, stream>>>(cw[6], cs[6], csh[6], lbl, lv); // conv7

    // ---- Segmentation head
    cat1280_kernel<<<((long)B_ * 1280 * N_ + 255) / 256, 256, 0, stream>>>(g1024, lv, x1, x2, x3, big);
    gemm_cbr(cw[7], big, h8,  cs[7], csh[7], 256, 1280, N_, 2, stream);     // conv8
    gemm_cbr(cw[8], h8,  h9,  cs[8], csh[8], 256,  256, N_, 2, stream);     // conv9
    gemm_cbr(cw[9], h9,  h10, cs[9], csh[9], 128,  256, N_, 2, stream);     // conv10
    final_kernel<<<(BN + 255) / 256, 256, 0, stream>>>(conv11_w, h10, out); // conv11 + tanh
}
